// EfficientAttention_48172353192059
// MI455X (gfx1250) — matrix operand order repaired and checked by reading, not compiled, not run on hardware
//
#include <hip/hip_runtime.h>
#include <hip/hip_bf16.h>

// Problem constants (fixed by the reference)
#define BATCH   4
#define CHAN    256
#define NSEQ    4096      // 64*64
#define HEADS   2
#define HDIM    16
#define RED     32        // reduced channels
#define NBLK    256       // NSEQ / 16
#define NT      256       // n-tile for projection kernels

typedef __attribute__((ext_vector_type(16))) _Float16 v16h;
typedef __attribute__((ext_vector_type(8)))  _Float16 v8h;
typedef __attribute__((ext_vector_type(8)))  float    v8f;

union V16U { v16h v; v8h h[2]; };

// Swap bits 3<->4 of a 5-bit lane/column index: reorders each 32-column group
// as [0..7,16..23,8..15,24..31] so a WMMA 16-bit A-operand (16x32) becomes one
// contiguous 32-byte load per lane.
__device__ __forceinline__ int swap34(int l) {
    return (l & 7) | ((l & 8) << 1) | ((l & 16) >> 1);
}

// ---------------------------------------------------------------------------
// Kernel 1: QKV 1x1-conv projection + BatchNorm(eval).
// Block = (b, n-tile of 256, one of q/k/v). Thread t owns column n = tile+t and
// keeps 32 accumulators in VGPRs; x is read once per c (coalesced), the 32
// weights per c are block-uniform -> scalar loads through the constant cache.
// q16,k16: [B,h,n,d] f16 (Q pre-scaled by d^-0.5); vT16: [B,h,d,n] f16 in
// natural n order (A-layout friendly).
// ---------------------------------------------------------------------------
__global__ void qkv_bn_kernel(const float* __restrict__ x,
                              const float* __restrict__ w_qkv,
                              const float* __restrict__ gamma,
                              const float* __restrict__ beta,
                              const float* __restrict__ mean,
                              const float* __restrict__ var,
                              _Float16* __restrict__ q16,
                              _Float16* __restrict__ k16,
                              _Float16* __restrict__ vT16)
{
    const int blk = blockIdx.x;
    const int og  = blk % 3;                       // 0=q, 1=k, 2=v
    const int nt  = (blk / 3) % (NSEQ / NT);
    const int b   = blk / (3 * (NSEQ / NT));
    const int t   = threadIdx.x;
    const int n   = nt * NT + t;
    const int nperm = n;  // vT store position

    const float* __restrict__ xb = x + (size_t)b * CHAN * NSEQ + n;
    const float* __restrict__ wg = w_qkv + (size_t)og * RED * CHAN; // uniform

    float acc[RED];
#pragma unroll
    for (int r = 0; r < RED; ++r) acc[r] = 0.f;

    for (int c = 0; c < CHAN; ++c) {
        float xv = xb[(size_t)c * NSEQ];
#pragma unroll
        for (int r = 0; r < RED; ++r)
            acc[r] = fmaf(wg[r * CHAN + c], xv, acc[r]);   // s_load weights
    }

#pragma unroll
    for (int r = 0; r < RED; ++r) {
        int o = og * RED + r;
        float istd = rsqrtf(var[o] + 1e-5f);
        float s = gamma[o] * istd;
        float v = acc[r] * s + (beta[o] - mean[o] * s);
        int hh = r >> 4, dd = r & 15;
        size_t bh = (size_t)(b * HEADS + hh);
        if (og == 0)      q16[(bh * NSEQ + n) * HDIM + dd] = (_Float16)(v * 0.25f);
        else if (og == 1) k16[(bh * NSEQ + n) * HDIM + dd] = (_Float16)v;
        else              vT16[(bh * HDIM + dd) * NSEQ + nperm] = (_Float16)v;
    }
}

// ---------------------------------------------------------------------------
// Kernel 2: flash attention, one wave32 per 16-row Q tile, 32 columns/iter.
//   S^T = K_tile (A) x Q^T (B)  (two 16-col tiles) -> softmax rows lane-local
//   O^T += V^T_tile (A, K=32) x P^T (B, K=32)      (one full-K wmma)
// clamp(-10,10) before softmax => fixed max of 10, no online-max bookkeeping;
// l_i cross-half reduction deferred to after the loop.
//
// QK A-operands: B rows K=16..31 (Q^T lanes 16-31) are zero, so the A upper
// K-half content is irrelevant (finite * 0 = 0). We therefore fill it with the
// adjacent real K data via a second clause-merged b128 load, letting loads
// write the 8-VGPR operand ranges in place (no zero re-materialization movs).
// ---------------------------------------------------------------------------
__global__ void flash_attn_kernel(const _Float16* __restrict__ q16,
                                  const _Float16* __restrict__ k16,
                                  const _Float16* __restrict__ vT16,
                                  float* __restrict__ attn_out)
{
    const int lane = threadIdx.x & 31;
    const int wave = threadIdx.x >> 5;
    const int tile = blockIdx.x * (blockDim.x >> 5) + wave; // 0..2047
    const int i  = tile & (NBLK - 1);         // q block index
    const int h  = (tile >> 8) & (HEADS - 1);
    const int b  = tile >> 9;
    const int m16 = lane & 15;
    const int hi  = lane >> 4;                // half-wave id

    const size_t bh = (size_t)(b * HEADS + h);
    const _Float16* qb = q16 + bh * NSEQ * HDIM;
    const _Float16* kb = k16 + bh * NSEQ * HDIM;
    const _Float16* vb = vT16 + (bh * HDIM + m16) * NSEQ;

    // Q^T as B-matrix: lanes 0-15 carry K=d=0..15 (full row), col = lane.
    // Lanes 16-31 (K=16..31) stay zero -> masks the A upper-half garbage.
    v16h bq = {};
    if (lane < 16)
        bq = *(const v16h*)(qb + (size_t)(i * 16 + m16) * HDIM);

    v8f Ot = {};            // O^T: row dd = g + 8*hi, col m = m16
    float l_i = 0.f;        // this half-wave's partial softmax denominator

    for (int j = 0; j < NSEQ; j += 32) {
        // Two K tiles as A-matrices (rows n = j+m16 and j+16+m16, lanes 0-15 hold d=0..15).
        // Lanes 16-31 (K=16..31 side) carry adjacent finite data, nullified by B=0.
        const _Float16* kp = kb + (size_t)(j + m16) * HDIM + hi * 8;
        V16U ak0, ak1;
        ak0.h[0] = *(const v8h*)(kp);
        ak0.h[1] = *(const v8h*)(kp + 8);
        ak1.h[0] = *(const v8h*)(kp + 16 * HDIM);
        ak1.h[1] = *(const v8h*)(kp + 16 * HDIM + 8);

        v8f cz = {};
        v8f St0 = __builtin_amdgcn_wmma_f32_16x16x32_f16(
            false, ak0.v, false, bq, (short)0, cz, false, false);
        v8f St1 = __builtin_amdgcn_wmma_f32_16x16x32_f16(
            false, ak1.v, false, bq, (short)0, cz, false, false);
        // St0[g]: S^T(n_local = g+8*hi,      m=m16); St1: n_local = 16+g+8*hi

        float p0[8], p1[8];
        float rs = 0.f;
#pragma unroll
        for (int g = 0; g < 8; ++g) {
            float s0 = fminf(10.f, fmaxf(-10.f, St0[g]));
            float s1 = fminf(10.f, fmaxf(-10.f, St1[g]));
            p0[g] = __expf(s0 - 10.f);
            p1[g] = __expf(s1 - 10.f);
            rs += p0[g] + p1[g];
        }
        l_i += rs;

        // P^T as B-matrix (K=32): lane(m,hi) elem e <-> n_local = e + 16*hi.
        // One swap-select + one xor-16 shuffle per VGPR serves both halves.
        V16U pb;
#pragma unroll
        for (int g = 0; g < 8; ++g) {
            float u = hi ? p0[g] : p1[g];       // value the partner half needs
            float w = __shfl_xor(u, 16, 32);    // hi=0 gets p0[g]; hi=1 gets p1[g]
            pb.v[g]     = (_Float16)(hi ? w : p0[g]);
            pb.v[g + 8] = (_Float16)(hi ? p1[g] : w);
        }

        // V^T tile as full K=32 A-matrix: one contiguous 32-byte load per lane
        // (vT16 in natural n order; k = 16*hi + s matches P^T's layout).
        V16U av;
        av.v = *(const v16h*)(vb + j + hi * 16);

        Ot = __builtin_amdgcn_wmma_f32_16x16x32_f16(
            false, av.v, false, pb.v, (short)0, Ot, false, false);
    }

    // Combine the two half-waves' partial denominators, normalize, store.
    l_i += __shfl_xor(l_i, 16, 32);
    float inv = 1.0f / l_i;
    const int nglob = i * 16 + m16;
    float* ob = attn_out + ((size_t)b * RED + h * HDIM) * NSEQ + nglob;
#pragma unroll
    for (int g = 0; g < 8; ++g)
        ob[(size_t)(g + 8 * hi) * NSEQ] = Ot[g] * inv;
}

// ---------------------------------------------------------------------------
// Kernel 3: out projection (C x RED) + residual:  y = x + alpha * W_out @ O.
// Block = (b, n-tile of 256, quarter of the 256 output channels). Thread t
// loads its 32 attn values into VGPRs once; w_out rows are block-uniform ->
// scalar loads. 32 FMA + 1 load + 1 store per output channel.
// ---------------------------------------------------------------------------
__global__ void out_proj_kernel(const float* __restrict__ x,
                                const float* __restrict__ w_out,
                                const float* __restrict__ alpha,
                                const float* __restrict__ attn,
                                float* __restrict__ y)
{
    const int blk = blockIdx.x;
    const int cq  = blk % 4;                       // quarter of channels
    const int nt  = (blk / 4) % (NSEQ / NT);
    const int b   = blk / (4 * (NSEQ / NT));
    const int t   = threadIdx.x;
    const int n   = nt * NT + t;

    float a[RED];
    const float* __restrict__ ab = attn + (size_t)b * RED * NSEQ + n;
#pragma unroll
    for (int r = 0; r < RED; ++r) a[r] = ab[(size_t)r * NSEQ];

    const float al = alpha[0];
    const int c0 = cq * (CHAN / 4);
    for (int c = c0; c < c0 + CHAN / 4; ++c) {
        const float* __restrict__ wc = w_out + (size_t)c * RED;  // uniform
        float acc = 0.f;
#pragma unroll
        for (int r = 0; r < RED; ++r)
            acc = fmaf(wc[r], a[r], acc);
        size_t idx = ((size_t)b * CHAN + c) * NSEQ + n;
        y[idx] = fmaf(al, acc, x[idx]);
    }
}

// ---------------------------------------------------------------------------
extern "C" void kernel_launch(void* const* d_in, const int* in_sizes, int n_in,
                              void* d_out, int out_size, void* d_ws, size_t ws_size,
                              hipStream_t stream)
{
    const float* x      = (const float*)d_in[0];
    const float* w_qkv  = (const float*)d_in[1];
    const float* gamma  = (const float*)d_in[2];
    const float* beta   = (const float*)d_in[3];
    const float* mean   = (const float*)d_in[4];
    const float* var    = (const float*)d_in[5];
    const float* w_out  = (const float*)d_in[6];
    const float* alpha  = (const float*)d_in[7];
    float* y = (float*)d_out;

    // Workspace: q16 (1MB) | k16 (1MB) | vT16 (1MB) | attn f32 (2MB)
    const size_t qkv_elems = (size_t)BATCH * HEADS * NSEQ * HDIM; // 524288
    _Float16* q16  = (_Float16*)d_ws;
    _Float16* k16  = q16 + qkv_elems;
    _Float16* vT16 = k16 + qkv_elems;
    float*    attn = (float*)(vT16 + qkv_elems);

    // (1) QKV projection + BN: B * (N/256) * 3 blocks
    qkv_bn_kernel<<<BATCH * (NSEQ / NT) * 3, NT, 0, stream>>>(
        x, w_qkv, gamma, beta, mean, var, q16, k16, vT16);

    // (2) flash attention: 2048 q-tiles, 8 waves per 256-thread block
    flash_attn_kernel<<<BATCH * HEADS * NBLK / 8, 256, 0, stream>>>(
        q16, k16, vT16, attn);

    // (3) out projection + residual: B * (N/256) * 4 blocks
    out_proj_kernel<<<BATCH * (NSEQ / NT) * 4, NT, 0, stream>>>(
        x, w_out, alpha, attn, y);
}
